// CTRGraphBlock_63986422775826
// MI455X (gfx1250) — matrix operand order repaired and checked by reading, not compiled, not run on hardware
//
#include <hip/hip_runtime.h>

// CTRGraphBlock fused pipeline for gfx1250 (MI455X), bf16 WMMA mixed precision.
// K0: mean over T  -> xm
// K1: q/k -> tanh -> relc, stored as pre-swizzled bf16 WMMA B-fragments
// K2: per (b, t-tile, Co-half): GEMM1 (v = Wv x, K=128, bf16 WMMA) into LDS,
//     GEMM2 (out += relc * v, K=Vpad32, bf16 WMMA) in register accumulators,
//     + GroupNorm partial sums
// K3: finalize group stats, K4: normalize + residual + relu

#define DEVINL __device__ __forceinline__

typedef __attribute__((ext_vector_type(16))) __bf16 bf16x16;
typedef __attribute__((ext_vector_type(8)))  float  f32x8;

constexpr int Bn = 64, Cc = 128, CoN = 128, Tn = 256, Vn = 25, Sn = 3, Rn = 16, Gn = 32;
constexpr int TILE_T  = 16;
constexpr int NFULL   = TILE_T * Vn;          // 400 = flattened (t,v) per tile
constexpr int NT      = 25;                   // 400 / 16 n-tiles
constexpr int XSTRIDE = 136;                  // padded ushort stride for xB[n][c]
constexpr int WSTRIDE = 136;                  // padded ushort stride for wv[o][c]
constexpr int VSTRIDE = 40;                   // padded v dim (>=32, 16B-aligned rows)

DEVINL unsigned short f2bf(float f) {         // fp32 -> bf16 RNE
  unsigned u = __float_as_uint(f);
  u += 0x7FFFu + ((u >> 16) & 1u);
  return (unsigned short)(u >> 16);
}

DEVINL f32x8 zero8() {
  f32x8 r;
#pragma unroll
  for (int i = 0; i < 8; ++i) r[i] = 0.f;
  return r;
}

DEVINL bf16x16 frag_2x8(const unsigned short* p0, const unsigned short* p1) {
  bf16x16 r;
  uint4* rv = reinterpret_cast<uint4*>(&r);
  rv[0] = *reinterpret_cast<const uint4*>(p0);
  rv[1] = *reinterpret_cast<const uint4*>(p1);
  return r;
}
DEVINL bf16x16 frag_16(const unsigned short* p) { return frag_2x8(p, p + 8); }

DEVINL f32x8 wmma_bf16(bf16x16 a, bf16x16 b, f32x8 c) {
  return __builtin_amdgcn_wmma_f32_16x16x32_bf16(false, a, false, b, (short)0, c,
                                                 false, false);
}

// ---------------- K0: xm = mean over T ----------------
__global__ __launch_bounds__(256)
void k_mean(const float* __restrict__ x, float* __restrict__ xm) {
  __shared__ float red[8][32];
  const int bc  = blockIdx.x;            // b*C + c
  const int tid = threadIdx.x;
  const int vI  = tid & 31, seg = tid >> 5;
  const float* p = x + (size_t)bc * Tn * Vn;
  float acc = 0.f;
  if (vI < Vn)
    for (int t = seg; t < Tn; t += 8) acc += p[t * Vn + vI];
  red[seg][vI] = acc;
  __syncthreads();
  if (tid < Vn) {
    float s2 = 0.f;
#pragma unroll
    for (int g = 0; g < 8; ++g) s2 += red[g][tid];
    xm[(size_t)bc * Vn + tid] = s2 * (1.f / (float)Tn);
  }
}

// ---------------- K1: relc as bf16 WMMA B-fragments ----------------
// Layout: relcB[b][s][o][ut][lane][16] : element j of lane holds
//   B[k = 16*(lane/16) + j][n = ut*16 + (lane%16)] = relc[b,s,o, u=n, v=k]
__global__ __launch_bounds__(256)
void k_relc(const float* __restrict__ xm, const float* __restrict__ Wq,
            const float* __restrict__ bq, const float* __restrict__ Wk,
            const float* __restrict__ bk, const float* __restrict__ Wr,
            const float* __restrict__ br, const float* __restrict__ Aadj,
            const float* __restrict__ alpha, unsigned short* __restrict__ relcB) {
  __shared__ float sb[16176];
  float* xm_s = sb;          // 3200  [c][u]
  float* q_s  = sb + 3200;   // 400   [r][u]
  float* k_s  = sb + 3600;   // 400   [r][v]
  float* wr_s = sb + 4000;   // 2048  [o][r]
  float* br_s = sb + 6048;   // 128
  float* trel = sb + 6176;   // 10000 [r][u][v]
  const int s = blockIdx.x, b = blockIdx.y, tid = threadIdx.x;

  for (int i = tid; i < Cc * Vn; i += 256) xm_s[i] = xm[(size_t)b * Cc * Vn + i];
  for (int i = tid; i < CoN * Rn; i += 256) wr_s[i] = Wr[(size_t)s * CoN * Rn + i];
  if (tid < CoN) br_s[tid] = br[s * CoN + tid];
  __syncthreads();

  for (int i = tid; i < 2 * Rn * Vn; i += 256) {
    const int isK = (i >= Rn * Vn) ? 1 : 0;
    const int rem = i - isK * Rn * Vn;
    const int r = rem / Vn, u = rem - r * Vn;
    const float* W = isK ? Wk : Wq;
    float acc = (isK ? bk : bq)[s * Rn + r];
    for (int c = 0; c < Cc; ++c)
      acc += W[((size_t)s * Rn + r) * Cc + c] * xm_s[c * Vn + u];
    (isK ? k_s : q_s)[rem] = acc;
  }
  __syncthreads();

  for (int i = tid; i < Rn * Vn * Vn; i += 256) {
    const int r = i / (Vn * Vn);
    const int rem = i - r * Vn * Vn;
    const int u = rem / Vn, v2 = rem - u * Vn;
    trel[i] = tanhf(q_s[r * Vn + u] - k_s[r * Vn + v2]);
  }
  __syncthreads();

  const float al = alpha[0];
  for (int row = tid; row < CoN * 2 * 32; row += 256) {
    const int o   = row >> 6;
    const int rem = row & 63;
    const int ut  = rem >> 5;
    const int ln  = rem & 31;
    const int hh  = ln >> 4;
    const int u   = ut * 16 + (ln & 15);
    float w16[16];
#pragma unroll
    for (int r = 0; r < 16; ++r) w16[r] = wr_s[o * 16 + r];
    const float bo = br_s[o];
    __align__(16) unsigned short tmp[16];
#pragma unroll
    for (int j = 0; j < 16; ++j) {
      const int kk = 16 * hh + j;  // v index
      float val = 0.f;
      if (u < Vn && kk < Vn) {
        float acc = bo;
#pragma unroll
        for (int r = 0; r < 16; ++r) acc += w16[r] * trel[r * (Vn * Vn) + u * Vn + kk];
        val = al * acc + Aadj[(size_t)s * Vn * Vn + u * Vn + kk];
      }
      tmp[j] = f2bf(val);
    }
    uint4* dst =
        reinterpret_cast<uint4*>(relcB + ((size_t)(b * Sn + s) * (CoN * 2 * 32) + row) * 16);
    dst[0] = *reinterpret_cast<uint4*>(tmp);
    dst[1] = *reinterpret_cast<uint4*>(tmp + 8);
  }
}

// ---------------- K2: main fused WMMA kernel ----------------
extern __shared__ unsigned char dynsmem[];

__global__ __launch_bounds__(256)
void k_main(const float* __restrict__ x, const float* __restrict__ Wv,
            const float* __restrict__ bv, const unsigned short* __restrict__ relcB,
            float* __restrict__ raw, float* __restrict__ part) {
  unsigned short* xB  = reinterpret_cast<unsigned short*>(dynsmem);  // [400][XSTRIDE]
  unsigned short* wvs = xB + NFULL * XSTRIDE;                        // [64][WSTRIDE]
  unsigned short* vls = wvs + 64 * WSTRIDE;                          // [64][16][VSTRIDE]
  float* gstat = reinterpret_cast<float*>(vls + 64 * 16 * VSTRIDE);  // [16][2]

  const int tid = threadIdx.x;
  const int lane = tid & 31;
  const int wid  = tid >> 5;
  const int h    = lane >> 4;
  const int l16  = lane & 15;
  const int b    = blockIdx.z;
  const int tt   = blockIdx.x;
  const int oh   = blockIdx.y;
  const int t0   = tt * TILE_T;
  const int ob0  = oh * 64;

  // zero padded v-region (pad columns stay 0 across all s iterations) + stats
  for (int i = tid; i < 64 * 16 * VSTRIDE; i += 256) vls[i] = 0;
  if (tid < 32) gstat[tid] = 0.f;

  // stage x tile transposed: xB[n][c] = bf16(x[b,c,t0 + n/25, n%25])
  const float* xb = x + (size_t)b * Cc * (Tn * Vn);
  for (int i = tid; i < Cc * NFULL; i += 256) {
    const int c = i / NFULL, n = i - c * NFULL;
    xB[n * XSTRIDE + c] = f2bf(xb[(size_t)c * (Tn * Vn) + t0 * Vn + n]);
  }
  __syncthreads();

  f32x8 acc2[8][2];
#pragma unroll
  for (int j = 0; j < 8; ++j) { acc2[j][0] = zero8(); acc2[j][1] = zero8(); }

  for (int s = 0; s < Sn; ++s) {
    // stage Wv tile as bf16 [o][c]
    for (int i = tid; i < 64 * Cc; i += 256) {
      const int o = i >> 7, c = i & 127;
      wvs[o * WSTRIDE + c] = f2bf(Wv[((size_t)s * CoN + ob0 + o) * Cc + c]);
    }
    __syncthreads();

    {  // GEMM1: v[o16, (t,v)400] = Wv[o16, c128] * x[c128, 400]   (2 waves / o-tile)
      const int ot = wid >> 1, hv = wid & 1;
      const int ntb = hv ? 13 : 0, nte = hv ? NT : 13;
      const unsigned short* arow = wvs + (ot * 16 + l16) * WSTRIDE + 16 * h;
      float bvv[8];
#pragma unroll
      for (int i2 = 0; i2 < 8; ++i2)
        bvv[i2] = bv[s * CoN + ob0 + ot * 16 + i2 + 8 * h];
      for (int nt = ntb; nt < nte; ++nt) {
        const int n = nt * 16 + l16;
        const unsigned short* brow = xB + n * XSTRIDE + 16 * h;
        f32x8 acc = zero8();
#pragma unroll
        for (int kb = 0; kb < Cc; kb += 32) {
          bf16x16 af = frag_16(arow + kb);                   // A: 16 contiguous K
          bf16x16 bf = frag_16(brow + kb);                   // B: 16 contiguous K
          acc = wmma_bf16(af, bf, acc);
        }
        const int tl = n / Vn, vvv = n - tl * Vn;
#pragma unroll
        for (int i2 = 0; i2 < 8; ++i2) {
          const int ob = ot * 16 + i2 + 8 * h;
          vls[(ob * 16 + tl) * VSTRIDE + vvv] = f2bf(acc[i2] + bvv[i2]);
        }
      }
    }
    __syncthreads();

    {  // GEMM2: out[t16, u32] += v[t16, v32] * relc[v32, u32]   (8 o per wave)
      const unsigned short* rb =
          relcB + ((((size_t)b * Sn + s) * CoN + ob0 + wid * 8) * 2) * (32 * 16);
#pragma unroll
      for (int j = 0; j < 8; ++j) {
        const int ob = wid * 8 + j;
        const unsigned short* ar = vls + (ob * 16 + l16) * VSTRIDE + 16 * h;
        bf16x16 af = frag_16(ar);
#pragma unroll
        for (int ut = 0; ut < 2; ++ut) {
          const unsigned short* bp = rb + ((size_t)(j * 2 + ut)) * (32 * 16) + lane * 16;
          __builtin_prefetch(bp + 1024, 0, 1);
          bf16x16 bfr = frag_16(bp);
          acc2[j][ut] = wmma_bf16(af, bfr, acc2[j][ut]);
        }
      }
    }
    __syncthreads();
  }

  // epilogue: store raw out + GroupNorm partials (group = 4 consecutive channels)
  float psum[2] = {0.f, 0.f}, psq[2] = {0.f, 0.f};
#pragma unroll
  for (int j = 0; j < 8; ++j) {
    const int og = ob0 + wid * 8 + j;
#pragma unroll
    for (int ut = 0; ut < 2; ++ut) {
      const int u = ut * 16 + l16;
      if (u < Vn) {
#pragma unroll
        for (int i2 = 0; i2 < 8; ++i2) {
          const int tl = i2 + 8 * h;
          const float val = acc2[j][ut][i2];
          raw[(((size_t)b * CoN + og) * Tn + t0 + tl) * Vn + u] = val;
          psum[j >> 2] += val;
          psq[j >> 2]  += val * val;
        }
      }
    }
  }
  atomicAdd(&gstat[(wid * 2 + 0) * 2 + 0], psum[0]);
  atomicAdd(&gstat[(wid * 2 + 0) * 2 + 1], psq[0]);
  atomicAdd(&gstat[(wid * 2 + 1) * 2 + 0], psum[1]);
  atomicAdd(&gstat[(wid * 2 + 1) * 2 + 1], psq[1]);
  __syncthreads();
  if (tid < 16) {
    const int g = oh * 16 + tid;
    part[(((size_t)b * Gn + g) * 16 + tt) * 2 + 0] = gstat[tid * 2 + 0];
    part[(((size_t)b * Gn + g) * 16 + tt) * 2 + 1] = gstat[tid * 2 + 1];
  }
}

// ---------------- K3: finalize group stats ----------------
__global__ void k_stats(const float* __restrict__ part, float* __restrict__ stat) {
  const int bg = blockIdx.x * blockDim.x + threadIdx.x;
  if (bg >= Bn * Gn) return;
  float sum = 0.f, sq = 0.f;
  for (int t = 0; t < 16; ++t) {
    sum += part[(size_t)bg * 32 + t * 2 + 0];
    sq  += part[(size_t)bg * 32 + t * 2 + 1];
  }
  const float inv = 1.f / (float)((CoN / Gn) * Tn * Vn);
  const float mu  = sum * inv;
  const float var = sq * inv - mu * mu;
  stat[bg * 2 + 0] = mu;
  stat[bg * 2 + 1] = rsqrtf(var + 1e-5f);
}

// ---------------- K4: normalize + affine + residual + relu ----------------
__global__ __launch_bounds__(256)
void k_final(const float* __restrict__ raw, const float* __restrict__ x,
             const float* __restrict__ stat, const float* __restrict__ gw,
             const float* __restrict__ gb, float* __restrict__ out) {
  const size_t idx = (size_t)blockIdx.x * 256 + threadIdx.x;
  const size_t total = (size_t)Bn * CoN * Tn * Vn;
  if (idx >= total) return;
  const int o = (int)((idx / (size_t)(Tn * Vn)) % CoN);
  const size_t bI = idx / ((size_t)CoN * Tn * Vn);
  const int g = o >> 2;
  const float mu = stat[(bI * Gn + g) * 2 + 0];
  const float rs = stat[(bI * Gn + g) * 2 + 1];
  const float val = (raw[idx] - mu) * rs * gw[o] + gb[o] + x[idx];
  out[idx] = fmaxf(val, 0.f);
}

extern "C" void kernel_launch(void* const* d_in, const int* in_sizes, int n_in,
                              void* d_out, int out_size, void* d_ws, size_t ws_size,
                              hipStream_t stream) {
  (void)in_sizes; (void)n_in; (void)out_size; (void)ws_size;
  const float* x     = (const float*)d_in[0];
  const float* Wq    = (const float*)d_in[1];
  const float* bq    = (const float*)d_in[2];
  const float* Wk    = (const float*)d_in[3];
  const float* bk    = (const float*)d_in[4];
  const float* Wv    = (const float*)d_in[5];
  const float* bv    = (const float*)d_in[6];
  const float* Wr    = (const float*)d_in[7];
  const float* br    = (const float*)d_in[8];
  const float* Aadj  = (const float*)d_in[9];
  const float* alpha = (const float*)d_in[10];
  const float* gnw   = (const float*)d_in[11];
  const float* gnb   = (const float*)d_in[12];

  char* ws = (char*)d_ws;
  size_t off = 0;
  float* xm = (float*)(ws + off);               off += (size_t)Bn * Cc * Vn * 4;
  unsigned short* relcB = (unsigned short*)(ws + off);
  off += (size_t)Bn * Sn * CoN * 2 * 32 * 16 * 2;
  float* raw  = (float*)(ws + off);             off += (size_t)Bn * CoN * Tn * Vn * 4;
  float* part = (float*)(ws + off);             off += (size_t)Bn * Gn * 16 * 2 * 4;
  float* stat = (float*)(ws + off);             off += (size_t)Bn * Gn * 2 * 4;

  k_mean<<<Bn * Cc, 256, 0, stream>>>(x, xm);
  k_relc<<<dim3(Sn, Bn), 256, 0, stream>>>(xm, Wq, bq, Wk, bk, Wr, br, Aadj, alpha, relcB);

  const size_t shmem =
      (size_t)(NFULL * XSTRIDE + 64 * WSTRIDE + 64 * 16 * VSTRIDE) * 2 + 32 * 4;
  (void)hipFuncSetAttribute((const void*)k_main,
                            hipFuncAttributeMaxDynamicSharedMemorySize, (int)shmem);
  k_main<<<dim3(Tn / TILE_T, 2, Bn), 256, shmem, stream>>>(x, Wv, bv, relcB, raw, part);

  k_stats<<<(Bn * Gn + 63) / 64, 64, 0, stream>>>(part, stat);

  const size_t total = (size_t)Bn * CoN * Tn * Vn;
  k_final<<<(unsigned)((total + 255) / 256), 256, 0, stream>>>(raw, x, stat, gnw, gnb,
                                                               (float*)d_out);
}
